// BitLinear_15822659519237
// MI455X (gfx1250) — matrix operand order repaired and checked by reading, not compiled, not run on hardware
//
#include <hip/hip_runtime.h>

// ---------------------------------------------------------------------------
// BitLinear (ternary-weight GEMM) for MI455X / gfx1250, wave32 + WMMA bf16.
//   out[8192,11008] = (x[8192,4096] * ternary(W)[11008,4096]^T) * scale
//   scale = mean(|W|) + 1e-5
// Roofline: 0.74 TFLOP vs 675 MB minimum HBM traffic -> memory-bound
// (~29us @ 23.3 TB/s). bf16 WMMA 16x16x32 (K=32/instr) keeps matrix math far
// below that floor; ternary W is exact in bf16. Fused on-the-fly quantization
// avoids any extra DRAM pass; 192MB L2 keeps W resident across the bm sweep.
// ---------------------------------------------------------------------------

typedef __attribute__((ext_vector_type(16))) __bf16   v16bf;
typedef __attribute__((ext_vector_type(2)))  __bf16   bf16x2;
typedef __attribute__((ext_vector_type(8)))  float    v8f;
typedef __attribute__((ext_vector_type(4)))  float    f32x4;
typedef __attribute__((ext_vector_type(4)))  unsigned u32x4;

#define GEMM_M 8192
#define GEMM_N 11008
#define GEMM_K 4096
#define BM 128
#define BN 128
#define BK 32
#define LDSTR 40          // padded LDS row stride in bf16 elems (80B, 16B aligned)
#define RED_BLOCKS 1024

// fp32 pair -> packed bf16x2 (native v_cvt path)
__device__ __forceinline__ unsigned pkbf(float f0, float f1) {
  bf16x2 h;
  h[0] = (__bf16)f0;
  h[1] = (__bf16)f1;
  return __builtin_bit_cast(unsigned, h);
}

union FragBF { u32x4 q[2]; v16bf v; };   // 32B: 16 bf16 = one WMMA A/B fragment

// ---------------- pass 1: per-block partial sums of |W| (deterministic) ----
__global__ __launch_bounds__(256)
void absmean_partial(const float* __restrict__ w, float* __restrict__ partials,
                     long long n) {
  __shared__ float sm[256];
  long long i = (long long)blockIdx.x * 256 + threadIdx.x;
  const long long stride = (long long)gridDim.x * 256;
  float s = 0.0f;
  for (; i < n; i += stride) s += fabsf(w[i]);
  sm[threadIdx.x] = s;
  __syncthreads();
  for (int o = 128; o > 0; o >>= 1) {
    if ((int)threadIdx.x < o) sm[threadIdx.x] += sm[threadIdx.x + o];
    __syncthreads();
  }
  if (threadIdx.x == 0) partials[blockIdx.x] = sm[0];
}

// ---------------- pass 2: final reduce -> scale, 1/scale -------------------
__global__ __launch_bounds__(256)
void absmean_final(const float* __restrict__ partials, float* __restrict__ scl,
                   float invN) {
  __shared__ float sm[256];
  float s = 0.0f;
  for (int i = threadIdx.x; i < RED_BLOCKS; i += 256) s += partials[i];
  sm[threadIdx.x] = s;
  __syncthreads();
  for (int o = 128; o > 0; o >>= 1) {
    if ((int)threadIdx.x < o) sm[threadIdx.x] += sm[threadIdx.x + o];
    __syncthreads();
  }
  if (threadIdx.x == 0) {
    float sc = sm[0] * invN + 1e-5f;
    scl[0] = sc;
    scl[1] = 1.0f / sc;
  }
}

// ---------------- fused quantize + WMMA GEMM -------------------------------
// 256 threads = 8 wave32. Block tile 128x128, K-step 32, double-buffered LDS.
// Wave grid 2(M) x 4(N): each wave computes 64x32 = 4x2 WMMA 16x16 tiles.
__global__ __launch_bounds__(256)
void bitlinear_wmma_gemm(const float* __restrict__ x, const float* __restrict__ w,
                         const float* __restrict__ scl, float* __restrict__ out) {
  __shared__ __align__(16) unsigned short As[2][BM * LDSTR];
  __shared__ __align__(16) unsigned short Bs[2][BN * LDSTR];

  const int tid  = threadIdx.x;
  const int lane = tid & 31;
  const int wid  = tid >> 5;
  const int wM   = (wid >> 2) * 64;      // 0 or 64
  const int wN   = (wid & 3) * 32;       // 0,32,64,96
  const int bm   = blockIdx.y * BM;
  const int bn   = blockIdx.x * BN;

  const float scale = scl[0];            // uniform (scalar) loads
  const float inv   = scl[1];

  // Tile loader: thread t handles row t/2, K-half (t&1)*16 -> 16 floats (4xb128)
  const int lr = tid >> 1;
  const int lh = (tid & 1) * 16;
  const float* ax = x + (size_t)(bm + lr) * GEMM_K + lh;
  const float* bw = w + (size_t)(bn + lr) * GEMM_K + lh;

  f32x4 ar[4], br[4];

  auto loadTile = [&](int k0) {
    const f32x4* pa = (const f32x4*)(ax + k0);
    const f32x4* pb = (const f32x4*)(bw + k0);
#pragma unroll
    for (int i = 0; i < 4; ++i) { ar[i] = pa[i]; br[i] = pb[i]; }
  };

  auto storeTile = [&](int buf) {
    unsigned ua[8], ub[8];
#pragma unroll
    for (int i = 0; i < 4; ++i) {
#pragma unroll
      for (int j = 0; j < 2; ++j) {
        // x -> bf16 (RNE via native cvt)
        ua[i * 2 + j] = pkbf(ar[i][j * 2], ar[i][j * 2 + 1]);
        // W -> ternary {-1,0,1}: round(W/scale) clamped via v_med3
        float q0 = __builtin_amdgcn_fmed3f(rintf(br[i][j * 2]     * inv), -1.0f, 1.0f);
        float q1 = __builtin_amdgcn_fmed3f(rintf(br[i][j * 2 + 1] * inv), -1.0f, 1.0f);
        ub[i * 2 + j] = pkbf(q0, q1);    // ternary values exact in bf16
      }
    }
    u32x4 t0 = {ua[0], ua[1], ua[2], ua[3]};
    u32x4 t1 = {ua[4], ua[5], ua[6], ua[7]};
    u32x4 t2 = {ub[0], ub[1], ub[2], ub[3]};
    u32x4 t3 = {ub[4], ub[5], ub[6], ub[7]};
    u32x4* da = (u32x4*)&As[buf][lr * LDSTR + lh];
    u32x4* db = (u32x4*)&Bs[buf][lr * LDSTR + lh];
    da[0] = t0; da[1] = t1;
    db[0] = t2; db[1] = t3;
  };

  v8f acc[4][2] = {};

  // CDNA5 fragment addressing (ISA 7.12.2):
  //  A 16x32 bf16: lanes 0-15 -> M=lane, K=0..7 & 16..23 ; lanes 16-31 -> K=8..15 & 24..31
  //  B 32x16 bf16: lanes 0-15 -> N=lane, K=0..7 & 16..23 ; lanes 16-31 -> K=8..15 & 24..31
  const int amBase = lane & 15;
  const int ak     = (lane & 16) ? 8  : 0;
  const int bk     = (lane & 16) ? 8  : 0;
  const int bRow0  = wN + (lane & 15);

  loadTile(0);
  storeTile(0);
  __syncthreads();

  const int NK = GEMM_K / BK;            // 128 K-steps
  for (int kt = 0; kt < NK; ++kt) {
    const int cur = kt & 1;
    if (kt + 1 < NK) loadTile((kt + 1) * BK);   // global prefetch overlaps WMMA

    const unsigned short* Ab = As[cur];
    const unsigned short* Bb = Bs[cur];

    // Issue ALL fragment ds_loads first so waits can be partial and LDS
    // latency overlaps the WMMA chain.
    FragBF fb[2];
    FragBF fa[4];
    {
      const unsigned short* p = Bb + (size_t)bRow0 * LDSTR + bk;
      fb[0].q[0] = *(const u32x4*)(p);
      fb[0].q[1] = *(const u32x4*)(p + 16);
      p += 16 * LDSTR;
      fb[1].q[0] = *(const u32x4*)(p);
      fb[1].q[1] = *(const u32x4*)(p + 16);
    }
#pragma unroll
    for (int mt = 0; mt < 4; ++mt) {
      const unsigned short* p = Ab + (size_t)(wM + mt * 16 + amBase) * LDSTR + ak;
      fa[mt].q[0] = *(const u32x4*)(p);
      fa[mt].q[1] = *(const u32x4*)(p + 16);
    }

#pragma unroll
    for (int mt = 0; mt < 4; ++mt) {
      acc[mt][0] = __builtin_amdgcn_wmma_f32_16x16x32_bf16(
          false, fa[mt].v, false, fb[0].v, (short)0, acc[mt][0], false, false);
      acc[mt][1] = __builtin_amdgcn_wmma_f32_16x16x32_bf16(
          false, fa[mt].v, false, fb[1].v, (short)0, acc[mt][1], false, false);
    }

    if (kt + 1 < NK) storeTile(1 - cur);
    __syncthreads();
  }

  // C/D layout: VGPR r -> M = r (lanes 0-15) / r+8 (lanes 16-31); N = lane&15
  const int mOff = (lane & 16) ? 8 : 0;
  const int nCol = lane & 15;
#pragma unroll
  for (int mt = 0; mt < 4; ++mt) {
#pragma unroll
    for (int nt = 0; nt < 2; ++nt) {
      const int n = bn + wN + nt * 16 + nCol;
#pragma unroll
      for (int r = 0; r < 8; ++r) {
        const int m = bm + wM + mt * 16 + mOff + r;
        out[(size_t)m * GEMM_N + n] = acc[mt][nt][r] * scale;
      }
    }
  }
}

// ---------------------------------------------------------------------------
extern "C" void kernel_launch(void* const* d_in, const int* in_sizes, int n_in,
                              void* d_out, int out_size, void* d_ws, size_t ws_size,
                              hipStream_t stream) {
  (void)in_sizes; (void)n_in; (void)out_size; (void)ws_size;

  const float* x   = (const float*)d_in[0];   // [4,2048,4096] -> [8192,4096]
  const float* w   = (const float*)d_in[1];   // [11008,4096]
  float*       out = (float*)d_out;           // [8192,11008]

  float* wsf      = (float*)d_ws;
  float* scl      = wsf;                      // [0]=scale, [1]=1/scale
  float* partials = wsf + 16;                 // RED_BLOCKS floats

  const long long nW = (long long)GEMM_N * GEMM_K;

  absmean_partial<<<RED_BLOCKS, 256, 0, stream>>>(w, partials, nW);
  absmean_final<<<1, 256, 0, stream>>>(partials, scl,
                                       (float)(1.0 / (double)nW));

  dim3 grid(GEMM_N / BN, GEMM_M / BM);        // (86, 64)
  bitlinear_wmma_gemm<<<grid, 256, 0, stream>>>(x, w, scl, out);
}
